// ScaledDotProductAttention_7129645711647
// MI455X (gfx1250) — matrix operand order repaired and checked by reading, not compiled, not run on hardware
//
#include <hip/hip_runtime.h>
#include <cstdint>

#define B_    16
#define TQ_   1024
#define TK_   1024
#define H_    512
#define NEGV  (-1.0e10f)
#define QTILE 32
#define KTILE 64
#define QSTR  520   // ushort stride (512 + 8 pad), rows 1040B (16B aligned)
#define KSTR  520
#define VTSTR 72    // sVt rows: 64 keys + 8 pad -> 144B rows (16B aligned, conflict-free loads)
#define PSTR  72

// scale * log2(e): softmax done in exp2 domain (v_exp_f32 is 2^x)
#define SCALE_L2 0.06375870739559747f
#define LOG2E    1.4426950408889634f

typedef __attribute__((ext_vector_type(16))) __bf16 v16bf;
typedef __attribute__((ext_vector_type(8)))  float  v8f;

// LDS layout (bytes)
#define SQ_OFF     0        // 32*520*2   = 33280
#define SK_OFF     33280    // 64*520*2   = 66560  (K as [key][h], feeds QK B-frag)
#define SVT_OFF    99840    // 512*72*2   = 73728  (V^T as [h][key], feeds PV B-frag)
#define SP_OFF     173568   // 32*72*2    = 4608
#define RMAX_OFF   178176   // 512
#define RSUM_OFF   178688   // 512
#define SM_OFF     179200   // 128
#define SL_OFF     179328   // 128
#define SMN_OFF    179456   // 128
#define SAL_OFF    179584   // 128
#define SMEM_BYTES 179712

// workspace layout (bytes)
#define WS_KBF_OFF 0u                  // 16*1024*512*2 = 16777216
#define WS_VT_OFF  16777216u           // 16*512*1024*2 = 16777216
#define WS_NEED    33554432u

__device__ __forceinline__ unsigned int pk2(float a, float b) {
  unsigned short x = __builtin_bit_cast(unsigned short, (__bf16)a);
  unsigned short y = __builtin_bit_cast(unsigned short, (__bf16)b);
  return (unsigned int)x | ((unsigned int)y << 16);
}
__device__ __forceinline__ __bf16 usbf(unsigned short u) {
  return __builtin_bit_cast(__bf16, u);
}
__device__ __forceinline__ v8f zero8() {
  v8f z;
#pragma unroll
  for (int i = 0; i < 8; ++i) z[i] = 0.0f;
  return z;
}

// ---------------- prepass 1: rowwise f32 -> bf16 (K) ----------------
__global__ __launch_bounds__(256) void cvt_bf16_rows(const float* __restrict__ src,
                                                     unsigned short* __restrict__ dst,
                                                     int n8) {
  int i = blockIdx.x * 256 + threadIdx.x;
  if (i >= n8) return;
  const float4* s = (const float4*)src + 2 * (size_t)i;
  float4 a = s[0], b = s[1];
  ((uint4*)dst)[i] = make_uint4(pk2(a.x, a.y), pk2(a.z, a.w),
                                pk2(b.x, b.y), pk2(b.z, b.w));
}

// ---------------- prepass 2: V^T, f32 [b][k][h] -> bf16 [b][h][k] ----------------
__global__ __launch_bounds__(256) void transpose_vt(const float* __restrict__ KV,
                                                    unsigned short* __restrict__ Vt) {
  __shared__ unsigned short tile[64][72];
  const int bid = blockIdx.x;            // 16 batches * 128 tiles
  const int b   = bid >> 7;
  const int t   = bid & 127;
  const int k0  = (t >> 3) * 64;
  const int h0  = (t & 7) * 64;
  const int tid = threadIdx.x;
  {
    int r = tid >> 4, c = tid & 15;
    const float* base = KV + ((size_t)b * TK_ + k0) * H_ + h0;
#pragma unroll
    for (int i = 0; i < 4; ++i) {
      int row = r + i * 16;
      float4 v = *(const float4*)(base + (size_t)row * H_ + c * 4);
      *(uint2*)(&tile[row][c * 4]) = make_uint2(pk2(v.x, v.y), pk2(v.z, v.w));
    }
  }
  __syncthreads();
  {
    int hr = tid >> 2, cq = tid & 3;
    unsigned short* obase = Vt + ((size_t)b * H_ + h0 + hr) * TK_ + k0;
#pragma unroll
    for (int i = 0; i < 2; ++i) {
      int kq = cq * 2 + i;
      unsigned int w0 = (unsigned int)tile[kq*8+0][hr] | ((unsigned int)tile[kq*8+1][hr] << 16);
      unsigned int w1 = (unsigned int)tile[kq*8+2][hr] | ((unsigned int)tile[kq*8+3][hr] << 16);
      unsigned int w2 = (unsigned int)tile[kq*8+4][hr] | ((unsigned int)tile[kq*8+5][hr] << 16);
      unsigned int w3 = (unsigned int)tile[kq*8+6][hr] | ((unsigned int)tile[kq*8+7][hr] << 16);
      *(uint4*)(obase + kq * 8) = make_uint4(w0, w1, w2, w3);
    }
  }
}

// ================= main flash-attention kernel (pre-converted K/V) =================
__global__ __launch_bounds__(256) void flash_attn_pre(
    const float* __restrict__ Q, const unsigned short* __restrict__ Kbf,
    const unsigned short* __restrict__ Vt, const int* __restrict__ lens,
    float* __restrict__ Out) {
  extern __shared__ char smem[];
  unsigned short* sQ  = (unsigned short*)(smem + SQ_OFF);
  unsigned short* sK  = (unsigned short*)(smem + SK_OFF);
  unsigned short* sVt = (unsigned short*)(smem + SVT_OFF);
  unsigned short* sP  = (unsigned short*)(smem + SP_OFF);
  float* sRmax = (float*)(smem + RMAX_OFF);
  float* sRsum = (float*)(smem + RSUM_OFF);
  float* sM    = (float*)(smem + SM_OFF);
  float* sL    = (float*)(smem + SL_OFF);
  float* sMn   = (float*)(smem + SMN_OFF);
  float* sAl   = (float*)(smem + SAL_OFF);

  const int tid  = threadIdx.x;
  const int w    = tid >> 5;
  const int lane = tid & 31;
  const int nl   = lane & 15;
  const int hi   = lane >> 4;

  const int blk = blockIdx.x;
  const int b   = blk >> 5;
  const int q0  = (blk & 31) * QTILE;
  const int len = lens[b];

  if (tid < 32) { sM[tid] = -INFINITY; sL[tid] = 0.0f; }

  // ---- stage Q tile (32 x 512) fp32 -> bf16 (once per block) ----
  {
    const float* qbase = Q + ((size_t)b * TQ_ + q0) * H_;
#pragma unroll
    for (int i = 0; i < 16; ++i) {
      int idx = tid + i * 256;
      int row = idx >> 7;
      int c4  = idx & 127;
      float4 v = ((const float4*)(qbase + (size_t)row * H_))[c4];
      *(uint2*)(sQ + row * QSTR + c4 * 4) = make_uint2(pk2(v.x, v.y), pk2(v.z, v.w));
    }
  }
  __syncthreads();

  const int mi = w >> 2;
  const int ni = w & 3;

  v8f acc[2][4];
#pragma unroll
  for (int a = 0; a < 2; ++a)
#pragma unroll
    for (int c = 0; c < 4; ++c) acc[a][c] = zero8();

  const int ktEnd = (len + KTILE - 1) / KTILE;

  for (int kt = 0; kt < ktEnd; ++kt) {
    // ---- stage K tile [64 x 512] bf16: pure b128 copies ----
    const unsigned short* kb = Kbf + ((size_t)b * TK_ + kt * KTILE) * H_;
#pragma unroll
    for (int i = 0; i < 16; ++i) {
      int idx = tid + i * 256;            // 4096 uint4
      int row = idx >> 6;                 // 64 uint4 per row
      int c16 = idx & 63;
      *(uint4*)(sK + row * KSTR + c16 * 8) = ((const uint4*)(kb + (size_t)row * H_))[c16];
    }
    // ---- stage V^T tile [512 x 64] bf16: pure b128 copies ----
    const unsigned short* vb = Vt + (size_t)b * H_ * TK_ + kt * KTILE;
#pragma unroll
    for (int i = 0; i < 16; ++i) {
      int idx = tid + i * 256;            // 4096 uint4
      int row = idx >> 3;                 // 8 uint4 per row (64 keys)
      int c16 = idx & 7;
      *(uint4*)(sVt + row * VTSTR + c16 * 8) = ((const uint4*)(vb + (size_t)row * TK_))[c16];
    }
    if (kt + 1 < ktEnd) {  // prefetch next K tile -> global_prefetch_b8
      const char* nb = (const char*)(kb + (size_t)KTILE * H_);
      __builtin_prefetch(nb + tid * 256, 0, 1);
    }
    __syncthreads();

    // ---- QK^T: wave computes score tile (mi, ni), K-dim = 512 ----
    v8f s = zero8();
    const unsigned short* aBase = sQ + (mi * 16 + nl) * QSTR;
    const unsigned short* bBase = sK + (ni * 16 + nl) * KSTR;
#pragma unroll
    for (int kk = 0; kk < H_; kk += 32) {
      v16bf a, bb;
#pragma unroll
      for (int j = 0; j < 8; ++j) {
        int kbA = ((j < 4) ? (2 * j) : (16 + 2 * (j - 4))) + hi * 8;
        int kbB = ((j < 4) ? (2 * j) : (16 + 2 * (j - 4))) + hi * 8;
        a[2 * j]      = usbf(aBase[kk + kbA]);
        a[2 * j + 1]  = usbf(aBase[kk + kbA + 1]);
        bb[2 * j]     = usbf(bBase[kk + kbB]);
        bb[2 * j + 1] = usbf(bBase[kk + kbB + 1]);
      }
      s = __builtin_amdgcn_wmma_f32_16x16x32_bf16(false, a, false, bb,
                                                  (short)0, s, false, false);
    }

    // ---- scale(log2 domain) + mask + per-wave row max ----
    const int key = kt * KTILE + ni * 16 + nl;
    float sv[8];
#pragma unroll
    for (int r = 0; r < 8; ++r) {
      sv[r] = (key < len) ? s[r] * SCALE_L2 : NEGV;
      float v = sv[r];
#pragma unroll
      for (int off = 8; off > 0; off >>= 1) v = fmaxf(v, __shfl_xor(v, off, 16));
      if (nl == 0) sRmax[(mi * 4 + ni) * 16 + r + 8 * hi] = v;
    }
    __syncthreads();

    if (tid < 32) {
      int m2 = tid >> 4, rl = tid & 15;
      float mo = sM[tid], mn = mo;
#pragma unroll
      for (int j = 0; j < 4; ++j) mn = fmaxf(mn, sRmax[(m2 * 4 + j) * 16 + rl]);
      sMn[tid] = mn;
      sAl[tid] = __builtin_amdgcn_exp2f(mo - mn);
      sM[tid]  = mn;
    }
    __syncthreads();

    // ---- P = 2^(sv - m_new), row sums, P tile -> LDS ----
#pragma unroll
    for (int r = 0; r < 8; ++r) {
      int row = mi * 16 + r + 8 * hi;
      float p = __builtin_amdgcn_exp2f(sv[r] - sMn[row]);
      sP[row * PSTR + ni * 16 + nl] = __builtin_bit_cast(unsigned short, (__bf16)p);
      float v = p;
#pragma unroll
      for (int off = 8; off > 0; off >>= 1) v += __shfl_xor(v, off, 16);
      if (nl == 0) sRsum[(mi * 4 + ni) * 16 + r + 8 * hi] = v;
    }
    __syncthreads();

    if (tid < 32) {
      int m2 = tid >> 4, rl = tid & 15;
      float t = 0.0f;
#pragma unroll
      for (int j = 0; j < 4; ++j) t += sRsum[(m2 * 4 + j) * 16 + rl];
      sL[tid] = sAl[tid] * sL[tid] + t;
    }

#pragma unroll
    for (int a = 0; a < 2; ++a)
#pragma unroll
      for (int r = 0; r < 8; ++r) {
        float al = sAl[a * 16 + r + 8 * hi];
#pragma unroll
        for (int c = 0; c < 4; ++c) acc[a][c][r] *= al;
      }

    // ---- PV ----
    const int colBase = w * 64;
#pragma unroll
    for (int kk2 = 0; kk2 < KTILE; kk2 += 32) {
      v16bf pa0, pa1;
      const unsigned short* p0 = sP + nl * PSTR + kk2;
      const unsigned short* p1 = sP + (16 + nl) * PSTR + kk2;
#pragma unroll
      for (int j = 0; j < 8; ++j) {
        int kbA = ((j < 4) ? (2 * j) : (16 + 2 * (j - 4))) + hi * 8;
        pa0[2 * j]     = usbf(p0[kbA]);
        pa0[2 * j + 1] = usbf(p0[kbA + 1]);
        pa1[2 * j]     = usbf(p1[kbA]);
        pa1[2 * j + 1] = usbf(p1[kbA + 1]);
      }
#pragma unroll
      for (int c = 0; c < 4; ++c) {
        v16bf vb2;
        const unsigned short* vB = sVt + (size_t)(colBase + c * 16 + nl) * VTSTR + kk2;
#pragma unroll
        for (int j = 0; j < 8; ++j) {
          int kbB = ((j < 4) ? (2 * j) : (16 + 2 * (j - 4))) + hi * 8;
          vb2[2 * j]     = usbf(vB[kbB]);
          vb2[2 * j + 1] = usbf(vB[kbB + 1]);
        }
        acc[0][c] = __builtin_amdgcn_wmma_f32_16x16x32_bf16(false, pa0, false, vb2,
                                                            (short)0, acc[0][c], false, false);
        acc[1][c] = __builtin_amdgcn_wmma_f32_16x16x32_bf16(false, pa1, false, vb2,
                                                            (short)0, acc[1][c], false, false);
      }
    }
    __syncthreads();
  }

  float* obase = Out + ((size_t)b * TQ_ + q0) * H_;
#pragma unroll
  for (int a = 0; a < 2; ++a)
#pragma unroll
    for (int r = 0; r < 8; ++r) {
      int row = a * 16 + r + 8 * hi;
      float inv = 1.0f / sL[row];
#pragma unroll
      for (int c = 0; c < 4; ++c)
        obase[(size_t)row * H_ + w * 64 + c * 16 + nl] = acc[a][c][r] * inv;
    }
}

// ================= fallback: fused kernel (no workspace) =================
__global__ __launch_bounds__(256) void flash_attn_fused(
    const float* __restrict__ Q, const float* __restrict__ KV,
    const int* __restrict__ lens, float* __restrict__ Out) {
  extern __shared__ char smem[];
  unsigned short* sQ  = (unsigned short*)(smem + SQ_OFF);
  unsigned short* sK  = (unsigned short*)(smem + SK_OFF);
  unsigned short* sVt = (unsigned short*)(smem + SVT_OFF);
  unsigned short* sP  = (unsigned short*)(smem + SP_OFF);
  float* sRmax = (float*)(smem + RMAX_OFF);
  float* sRsum = (float*)(smem + RSUM_OFF);
  float* sM    = (float*)(smem + SM_OFF);
  float* sL    = (float*)(smem + SL_OFF);
  float* sMn   = (float*)(smem + SMN_OFF);
  float* sAl   = (float*)(smem + SAL_OFF);

  const int tid  = threadIdx.x;
  const int w    = tid >> 5;
  const int lane = tid & 31;
  const int nl   = lane & 15;
  const int hi   = lane >> 4;

  const int blk = blockIdx.x;
  const int b   = blk >> 5;
  const int q0  = (blk & 31) * QTILE;
  const int len = lens[b];

  if (tid < 32) { sM[tid] = -INFINITY; sL[tid] = 0.0f; }

  {
    const float* qbase = Q + ((size_t)b * TQ_ + q0) * H_;
#pragma unroll
    for (int i = 0; i < 16; ++i) {
      int idx = tid + i * 256;
      int row = idx >> 7;
      int c4  = idx & 127;
      float4 v = ((const float4*)(qbase + (size_t)row * H_))[c4];
      *(uint2*)(sQ + row * QSTR + c4 * 4) = make_uint2(pk2(v.x, v.y), pk2(v.z, v.w));
    }
  }
  __syncthreads();

  const int mi = w >> 2;
  const int ni = w & 3;

  v8f acc[2][4];
#pragma unroll
  for (int a = 0; a < 2; ++a)
#pragma unroll
    for (int c = 0; c < 4; ++c) acc[a][c] = zero8();

  const int ktEnd = (len + KTILE - 1) / KTILE;

  for (int kt = 0; kt < ktEnd; ++kt) {
    const float* kbase = KV + ((size_t)b * TK_ + kt * KTILE) * H_;
#pragma unroll
    for (int i = 0; i < 8; ++i) {
      int idx = tid + i * 256;
      int kb4 = idx >> 7;
      int h4  = idx & 127;
      const float* gp = kbase + (size_t)(kb4 * 4) * H_ + h4 * 4;
      float4 r0 = *(const float4*)(gp);
      float4 r1 = *(const float4*)(gp + H_);
      float4 r2 = *(const float4*)(gp + 2 * H_);
      float4 r3 = *(const float4*)(gp + 3 * H_);
      unsigned short* kp = sK + (kb4 * 4) * KSTR + h4 * 4;
      *(uint2*)(kp)            = make_uint2(pk2(r0.x, r0.y), pk2(r0.z, r0.w));
      *(uint2*)(kp + KSTR)     = make_uint2(pk2(r1.x, r1.y), pk2(r1.z, r1.w));
      *(uint2*)(kp + 2 * KSTR) = make_uint2(pk2(r2.x, r2.y), pk2(r2.z, r2.w));
      *(uint2*)(kp + 3 * KSTR) = make_uint2(pk2(r3.x, r3.y), pk2(r3.z, r3.w));
      unsigned short* vp = sVt + (h4 * 4) * VTSTR + kb4 * 4;
      *(uint2*)(vp)             = make_uint2(pk2(r0.x, r1.x), pk2(r2.x, r3.x));
      *(uint2*)(vp + VTSTR)     = make_uint2(pk2(r0.y, r1.y), pk2(r2.y, r3.y));
      *(uint2*)(vp + 2 * VTSTR) = make_uint2(pk2(r0.z, r1.z), pk2(r2.z, r3.z));
      *(uint2*)(vp + 3 * VTSTR) = make_uint2(pk2(r0.w, r1.w), pk2(r2.w, r3.w));
    }
    __syncthreads();

    v8f s = zero8();
    const unsigned short* aBase = sQ + (mi * 16 + nl) * QSTR;
    const unsigned short* bBase = sK + (ni * 16 + nl) * KSTR;
#pragma unroll
    for (int kk = 0; kk < H_; kk += 32) {
      v16bf a, bb;
#pragma unroll
      for (int j = 0; j < 8; ++j) {
        int kbA = ((j < 4) ? (2 * j) : (16 + 2 * (j - 4))) + hi * 8;
        int kbB = ((j < 4) ? (2 * j) : (16 + 2 * (j - 4))) + hi * 8;
        a[2 * j]      = usbf(aBase[kk + kbA]);
        a[2 * j + 1]  = usbf(aBase[kk + kbA + 1]);
        bb[2 * j]     = usbf(bBase[kk + kbB]);
        bb[2 * j + 1] = usbf(bBase[kk + kbB + 1]);
      }
      s = __builtin_amdgcn_wmma_f32_16x16x32_bf16(false, a, false, bb,
                                                  (short)0, s, false, false);
    }

    const int key = kt * KTILE + ni * 16 + nl;
    float sv[8];
#pragma unroll
    for (int r = 0; r < 8; ++r) {
      sv[r] = (key < len) ? s[r] * SCALE_L2 : NEGV;
      float v = sv[r];
#pragma unroll
      for (int off = 8; off > 0; off >>= 1) v = fmaxf(v, __shfl_xor(v, off, 16));
      if (nl == 0) sRmax[(mi * 4 + ni) * 16 + r + 8 * hi] = v;
    }
    __syncthreads();

    if (tid < 32) {
      int m2 = tid >> 4, rl = tid & 15;
      float mo = sM[tid], mn = mo;
#pragma unroll
      for (int j = 0; j < 4; ++j) mn = fmaxf(mn, sRmax[(m2 * 4 + j) * 16 + rl]);
      sMn[tid] = mn;
      sAl[tid] = __builtin_amdgcn_exp2f(mo - mn);
      sM[tid]  = mn;
    }
    __syncthreads();

#pragma unroll
    for (int r = 0; r < 8; ++r) {
      int row = mi * 16 + r + 8 * hi;
      float p = __builtin_amdgcn_exp2f(sv[r] - sMn[row]);
      sP[row * PSTR + ni * 16 + nl] = __builtin_bit_cast(unsigned short, (__bf16)p);
      float v = p;
#pragma unroll
      for (int off = 8; off > 0; off >>= 1) v += __shfl_xor(v, off, 16);
      if (nl == 0) sRsum[(mi * 4 + ni) * 16 + r + 8 * hi] = v;
    }
    __syncthreads();

    if (tid < 32) {
      int m2 = tid >> 4, rl = tid & 15;
      float t = 0.0f;
#pragma unroll
      for (int j = 0; j < 4; ++j) t += sRsum[(m2 * 4 + j) * 16 + rl];
      sL[tid] = sAl[tid] * sL[tid] + t;
    }

#pragma unroll
    for (int a = 0; a < 2; ++a)
#pragma unroll
      for (int r = 0; r < 8; ++r) {
        float al = sAl[a * 16 + r + 8 * hi];
#pragma unroll
        for (int c = 0; c < 4; ++c) acc[a][c][r] *= al;
      }

    const int colBase = w * 64;
#pragma unroll
    for (int kk2 = 0; kk2 < KTILE; kk2 += 32) {
      v16bf pa0, pa1;
      const unsigned short* p0 = sP + nl * PSTR + kk2;
      const unsigned short* p1 = sP + (16 + nl) * PSTR + kk2;
#pragma unroll
      for (int j = 0; j < 8; ++j) {
        int kbA = ((j < 4) ? (2 * j) : (16 + 2 * (j - 4))) + hi * 8;
        pa0[2 * j]     = usbf(p0[kbA]);
        pa0[2 * j + 1] = usbf(p0[kbA + 1]);
        pa1[2 * j]     = usbf(p1[kbA]);
        pa1[2 * j + 1] = usbf(p1[kbA + 1]);
      }
#pragma unroll
      for (int c = 0; c < 4; ++c) {
        v16bf vb2;
        const unsigned short* vB = sVt + (size_t)(colBase + c * 16 + nl) * VTSTR + kk2;
#pragma unroll
        for (int j = 0; j < 8; ++j) {
          int kbB = ((j < 4) ? (2 * j) : (16 + 2 * (j - 4))) + hi * 8;
          vb2[2 * j]     = usbf(vB[kbB]);
          vb2[2 * j + 1] = usbf(vB[kbB + 1]);
        }
        acc[0][c] = __builtin_amdgcn_wmma_f32_16x16x32_bf16(false, pa0, false, vb2,
                                                            (short)0, acc[0][c], false, false);
        acc[1][c] = __builtin_amdgcn_wmma_f32_16x16x32_bf16(false, pa1, false, vb2,
                                                            (short)0, acc[1][c], false, false);
      }
    }
    __syncthreads();
  }

  float* obase = Out + ((size_t)b * TQ_ + q0) * H_;
#pragma unroll
  for (int a = 0; a < 2; ++a)
#pragma unroll
    for (int r = 0; r < 8; ++r) {
      int row = a * 16 + r + 8 * hi;
      float inv = 1.0f / sL[row];
#pragma unroll
      for (int c = 0; c < 4; ++c)
        obase[(size_t)row * H_ + w * 64 + c * 16 + nl] = acc[a][c][r] * inv;
    }
}

extern "C" void kernel_launch(void* const* d_in, const int* in_sizes, int n_in,
                              void* d_out, int out_size, void* d_ws, size_t ws_size,
                              hipStream_t stream) {
  (void)in_sizes; (void)n_in; (void)out_size;
  const float* Q    = (const float*)d_in[0];
  const float* KV   = (const float*)d_in[1];
  const int*   lens = (const int*)d_in[2];
  float*       Out  = (float*)d_out;

  dim3 block(256);
  if (ws_size >= (size_t)WS_NEED && d_ws != nullptr) {
    unsigned short* Kbf = (unsigned short*)((char*)d_ws + WS_KBF_OFF);
    unsigned short* Vt  = (unsigned short*)((char*)d_ws + WS_VT_OFF);
    // prepass: K -> bf16 rows; V -> bf16 transposed
    hipLaunchKernelGGL(cvt_bf16_rows, dim3(4096), block, 0, stream,
                       KV, Kbf, (int)(B_ * TK_ * H_ / 8));
    hipLaunchKernelGGL(transpose_vt, dim3(2048), block, 0, stream, KV, Vt);
    hipFuncSetAttribute((const void*)flash_attn_pre,
                        hipFuncAttributeMaxDynamicSharedMemorySize, SMEM_BYTES);
    hipLaunchKernelGGL(flash_attn_pre, dim3(B_ * (TQ_ / QTILE)), block, SMEM_BYTES,
                       stream, Q, Kbf, Vt, lens, Out);
  } else {
    hipFuncSetAttribute((const void*)flash_attn_fused,
                        hipFuncAttributeMaxDynamicSharedMemorySize, SMEM_BYTES);
    hipLaunchKernelGGL(flash_attn_fused, dim3(B_ * (TQ_ / QTILE)), block, SMEM_BYTES,
                       stream, Q, KV, lens, Out);
  }
}
